// Quantizer_51634096832515
// MI455X (gfx1250) — matrix operand order repaired and checked by reading, not compiled, not run on hardware
//
#include <hip/hip_runtime.h>

// ---------------------------------------------------------------------------
// VQ-VAE quantizer for MI455X (gfx1250, wave32, WMMA).
//   z:   [32,1024,256] f32  -> N=32768 rows, D=256
//   emb: [8192,256]    f32  -> K=8192 codes
// argmin_k( ||e_k||^2 - 2 z.e_k )  (the ||z||^2 term is row-constant)
// Dominant cost: 32768 x 8192 x 256 GEMM = 137 GFLOP -> V_WMMA_F32_16X16X32_F16
// (f16 in, f32 accumulate; compute-bound, AI ~1900 FLOP/B). All other kernels
// are ~100 MB of bandwidth (~4.5 us at 23.3 TB/s).
// B tiles are staged to LDS with GLOBAL_LOAD_ASYNC_TO_LDS_B128 (ASYNCcnt)
// when the toolchain exposes the builtin; plain ld+ds_store otherwise.
// ---------------------------------------------------------------------------

typedef __attribute__((ext_vector_type(16))) _Float16 v16h;
typedef __attribute__((ext_vector_type(8)))  float    v8f;
typedef int v4i __attribute__((vector_size(16)));   // matches builtin param type

#define DDIM    256
#define KEMB    8192
#define NROWS   32768
#define CHUNKS  (DDIM / 32)    // 8 k-chunks of 32 per WMMA
#define NTILES  (KEMB / 16)    // 512 column tiles
#define RTILES  (NROWS / 16)   // 2048 row tiles

#if __has_builtin(__builtin_amdgcn_global_load_async_to_lds_b128)
#define USE_ASYNC_LDS 1
#endif

#ifdef USE_ASYNC_LDS
// Copy 32 bytes (one v16h fragment) global -> LDS without touching VGPRs.
// Tracked by ASYNCcnt; consumer must s_wait_asynccnt + barrier.
__device__ __forceinline__ void stage32_async(const v16h* g, v16h* l) {
  v4i* gp = (v4i*)g;
  v4i* lp = (v4i*)l;
  __builtin_amdgcn_global_load_async_to_lds_b128(gp, lp, 0, 0);
  __builtin_amdgcn_global_load_async_to_lds_b128(gp, lp, 16, 0);
}
#endif

// --------------------------------------------------------------------------
// Kernel 1: z (f32) -> A-fragment-swizzled f16.
// 16x32 f16 A order used here (contiguous k): lane L holds row M = L%16; element e holds
//   K = chunk*32 + (L/16)*16 + e.
// Ap element index (as v16h): (rowTile*8 + chunk)*32 + lane.
// --------------------------------------------------------------------------
__global__ __launch_bounds__(256) void prep_z(const float* __restrict__ z,
                                              _Float16* __restrict__ Ap) {
  const int rt    = blockIdx.x;          // row tile (16 rows)
  const int t     = threadIdx.x;         // 256 threads = 8 chunks x 32 lanes
  const int lane  = t & 31;
  const int chunk = t >> 5;
  const int m     = rt * 16 + (lane & 15);
  const int kbase = chunk * 32 + ((lane >> 4) << 4);
  const float* __restrict__ src = z + (size_t)m * DDIM;
  v16h frag;
#pragma unroll
  for (int e = 0; e < 16; ++e) {
    const int k = kbase + e;
    frag[e] = (_Float16)src[k];
  }
  *((v16h*)Ap + (size_t)(rt * 8 + chunk) * 32 + lane) = frag;
}

// --------------------------------------------------------------------------
// Kernel 2: embedding (f32) -> B-fragment-swizzled f16.
// 32x16 f16 B order used here (same contiguous k as A): lane L holds column N = L%16; element e holds
//   K = chunk*32 + (L/16)*16 + e.
// --------------------------------------------------------------------------
__global__ __launch_bounds__(256) void prep_e(const float* __restrict__ emb,
                                              _Float16* __restrict__ Bp) {
  const int tile  = blockIdx.x;          // column tile (16 codes)
  const int t     = threadIdx.x;
  const int lane  = t & 31;
  const int chunk = t >> 5;
  const int n     = tile * 16 + (lane & 15);
  const int kbase = chunk * 32 + ((lane >> 4) << 4);
  const float* __restrict__ src = emb + (size_t)n * DDIM + kbase;
  v16h frag;
#pragma unroll
  for (int e = 0; e < 16; ++e) frag[e] = (_Float16)src[e];
  *((v16h*)Bp + (size_t)(tile * 8 + chunk) * 32 + lane) = frag;
}

// --------------------------------------------------------------------------
// Kernel 3: exact f32 ||e_k||^2  (one block per code)
// --------------------------------------------------------------------------
__global__ __launch_bounds__(256) void enorm_k(const float* __restrict__ emb,
                                               float* __restrict__ enorm) {
  __shared__ float red[256];
  const int k = blockIdx.x;
  const float v = emb[(size_t)k * DDIM + threadIdx.x];
  red[threadIdx.x] = v * v;
  __syncthreads();
#pragma unroll
  for (int s = 128; s > 0; s >>= 1) {
    if (threadIdx.x < s) red[threadIdx.x] += red[threadIdx.x + s];
    __syncthreads();
  }
  if (threadIdx.x == 0) enorm[k] = red[0];
}

// --------------------------------------------------------------------------
// Kernel 4: GEMM + argmin.
// Block = 256 threads = 8 waves; wave w owns row-tile rt = blockIdx*8 + w
// (16 rows, A kept in 64 VGPRs). Loop over 512 column tiles: B tile (8 KB)
// double-buffered in LDS, shared by all 8 waves (8x global-traffic cut).
// 8 x v_wmma per tile on TWO accumulator chains to halve XDL RAW stalls.
// C/D layout: lane L, vgpr r -> M = (L/16)*8 + r, N = L%16.
// --------------------------------------------------------------------------
__global__ __launch_bounds__(256) void vq_gemm_argmin(
    const _Float16* __restrict__ Ap, const _Float16* __restrict__ Bp,
    const float* __restrict__ enorm, int* __restrict__ outIdx) {
  __shared__ v16h lsB[2][256];           // 2 x 8 KB double buffer

  const int t    = threadIdx.x;
  const int lane = t & 31;
  const int wave = t >> 5;               // 0..7
  const int rt   = blockIdx.x * 8 + wave;

  // Persistent A fragments for this wave's 16 rows (8 chunks x v16h).
  const v16h* __restrict__ Af = (const v16h*)Ap + (size_t)rt * 8 * 32 + lane;
  v16h a[CHUNKS];
#pragma unroll
  for (int c = 0; c < CHUNKS; ++c) a[c] = Af[c * 32];

  float best[8];
  int   bidx[8];
#pragma unroll
  for (int r = 0; r < 8; ++r) { best[r] = 3.4028235e38f; bidx[r] = 0; }

  const v16h* __restrict__ Bf = (const v16h*)Bp;  // index: tile*256 + t

  // Prologue: stage tile 0.
#ifdef USE_ASYNC_LDS
  stage32_async(Bf + t, &lsB[0][t]);
  asm volatile("s_wait_asynccnt 0" ::: "memory");
#else
  lsB[0][t] = Bf[t];
#endif
  __syncthreads();

  const int ncol = lane & 15;
  for (int tile = 0; tile < NTILES; ++tile) {
    const int buf = tile & 1;
    // Stage next tile into the other buffer (overlaps with WMMA work).
    if (tile + 1 < NTILES) {
#ifdef USE_ASYNC_LDS
      stage32_async(Bf + (size_t)(tile + 1) * 256 + t, &lsB[buf ^ 1][t]);
#else
      lsB[buf ^ 1][t] = Bf[(size_t)(tile + 1) * 256 + t];
#endif
    }
    if (tile + 2 < NTILES) __builtin_prefetch(Bf + (size_t)(tile + 2) * 256, 0, 0);

    v8f acc0 = {0.f, 0.f, 0.f, 0.f, 0.f, 0.f, 0.f, 0.f};
    v8f acc1 = {0.f, 0.f, 0.f, 0.f, 0.f, 0.f, 0.f, 0.f};
#pragma unroll
    for (int c = 0; c < 4; ++c) {
      const v16h b0 = lsB[buf][c * 32 + lane];
      const v16h b1 = lsB[buf][(c + 4) * 32 + lane];
      acc0 = __builtin_amdgcn_wmma_f32_16x16x32_f16(
          false, a[c],     false, b0, (short)0, acc0, false, false);
      acc1 = __builtin_amdgcn_wmma_f32_16x16x32_f16(
          false, a[c + 4], false, b1, (short)0, acc1, false, false);
    }

    // score = ||e||^2 - 2 * (z . e);  per-lane running argmin (indices
    // ascend with tile, so strict < keeps the first minimum).
    const float en = enorm[tile * 16 + ncol];
    const int   n  = tile * 16 + ncol;
#pragma unroll
    for (int r = 0; r < 8; ++r) {
      const float s = fmaf(-2.0f, acc0[r] + acc1[r], en);
      if (s < best[r]) { best[r] = s; bidx[r] = n; }
    }

#ifdef USE_ASYNC_LDS
    asm volatile("s_wait_asynccnt 0" ::: "memory");  // buf^1 fully landed
#endif
    __syncthreads();   // producer/consumer handoff for LDS double buffer
  }

  // Cross-lane reduction: lane L's candidates cover columns = L%16 (mod 16)
  // for rows M = (L/16)*8 + r. Reduce the 16 column classes per row via LDS
  // (reusing the 16 KB staging buffer: 8 KB scores + 8 KB indices).
  __syncthreads();
  float* rs = (float*)&lsB[0][0];
  int*   ri = (int*)((char*)&lsB[0][0] + 8 * 256 * sizeof(float));
  const int h = lane >> 4;
#pragma unroll
  for (int r = 0; r < 8; ++r) {
    const int m    = h * 8 + r;
    const int slot = wave * 256 + m * 16 + ncol;
    rs[slot] = best[r];
    ri[slot] = bidx[r];
  }
  __syncthreads();
  if (lane < 16) {
    const int m = lane;
    float b = 3.4028235e38f;
    int   bi = 0x7fffffff;
    #pragma unroll
    for (int c = 0; c < 16; ++c) {
      const float s  = rs[wave * 256 + m * 16 + c];
      const int   id = ri[wave * 256 + m * 16 + c];
      if (s < b || (s == b && id < bi)) { b = s; bi = id; }
    }
    outIdx[rt * 16 + m] = bi;
  }
}

// --------------------------------------------------------------------------
// Kernel 5: gather original f32 codes + emit indices (as float) after z_q.
// --------------------------------------------------------------------------
__global__ __launch_bounds__(256) void gather_out(
    const float* __restrict__ emb, const int* __restrict__ idx,
    float* __restrict__ outQ, float* __restrict__ outI) {
  const int row = blockIdx.x;
  const int id  = idx[row];
  outQ[(size_t)row * DDIM + threadIdx.x] = emb[(size_t)id * DDIM + threadIdx.x];
  if (threadIdx.x == 0) outI[row] = (float)id;
}

// --------------------------------------------------------------------------
extern "C" void kernel_launch(void* const* d_in, const int* in_sizes, int n_in,
                              void* d_out, int out_size, void* d_ws, size_t ws_size,
                              hipStream_t stream) {
  const float* z   = (const float*)d_in[0];   // [32,1024,256] f32
  const float* emb = (const float*)d_in[1];   // [8192,256]    f32

  // Workspace layout (needs ~20.4 MB):
  //   Ap: swizzled f16 z        16 MB
  //   Bp: swizzled f16 codes     4 MB
  //   enorm: f32[8192]          32 KB
  //   idx: int32[32768]        128 KB
  char* ws = (char*)d_ws;
  _Float16* Ap    = (_Float16*)ws;
  _Float16* Bp    = (_Float16*)(ws + (size_t)NROWS * DDIM * sizeof(_Float16));
  float*    enorm = (float*)(ws + (size_t)NROWS * DDIM * sizeof(_Float16)
                                + (size_t)KEMB * DDIM * sizeof(_Float16));
  int*      idx   = (int*)((char*)enorm + (size_t)KEMB * sizeof(float));

  float* outQ = (float*)d_out;                 // [32768,256] f32
  float* outI = outQ + (size_t)NROWS * DDIM;   // [32768] indices (as f32)

  prep_z        <<<RTILES,     256, 0, stream>>>(z, Ap);
  prep_e        <<<NTILES,     256, 0, stream>>>(emb, Bp);
  enorm_k       <<<KEMB,       256, 0, stream>>>(emb, enorm);
  vq_gemm_argmin<<<RTILES / 8, 256, 0, stream>>>(Ap, Bp, enorm, idx);
  gather_out    <<<NROWS,      256, 0, stream>>>(emb, idx, outQ, outI);
}
